// GraphConvolutionLayer_26070451487089
// MI455X (gfx1250) — compile-verified, matrix operand order checked by reading, not run on hardware
//
#include <hip/hip_runtime.h>
#include <hip/hip_bf16.h>
#include <math.h>

typedef __attribute__((ext_vector_type(16))) __bf16 v16bf;
typedef __attribute__((ext_vector_type(8)))  float  v8f;

#define NN      100000
#define NE      1600000
#define IN_DIM  128
#define OUT_DIM 64
#define HEADS   4
#define HC      256          // HEADS*OUT_DIM
#define TOT_E   (NE + NN)    // edges + self loops
#define NEG_SLOPE 0.2f
#define LN_EPS  1e-5f

// ---------------------------------------------------------------- utilities
__device__ __forceinline__ void atomicMaxFloat(float* addr, float val) {
    // order-preserving int trick; amax initialized to -inf (0xFF800000)
    if (!(val < 0.0f)) atomicMax((int*)addr, __float_as_int(val));
    else               atomicMin((unsigned int*)addr, __float_as_uint(val));
}

__device__ __forceinline__ float lrelu(float a) {
    return a > 0.0f ? a : NEG_SLOPE * a;
}

// ---------------------------------------------------------------- init
__global__ void gat_init_kernel(float* __restrict__ accum,
                                float* __restrict__ amax,
                                float* __restrict__ denom) {
    int i = blockIdx.x * blockDim.x + threadIdx.x;
    if (i < NN * OUT_DIM) accum[i] = 0.0f;
    if (i < NN * HEADS) {
        ((int*)amax)[i] = 0xFF800000;   // -inf
        denom[i] = 0.0f;
    }
}

// ---------------------------------------------------------------- GEMM: xl = bf16(x @ W)
// block = 256 threads = 8 waves; block handles one 16-row M stripe.
// wave w computes N-tiles {2w, 2w+1} (16 tiles * 16 cols = 256 = full N).
__global__ void __launch_bounds__(256)
gat_gemm_kernel(const float* __restrict__ x,     // (NN,128)
                const float* __restrict__ W,     // (128,256)
                __bf16* __restrict__ xl) {       // (NN,256) bf16
    const int lane  = threadIdx.x & 31;
    const int wave  = threadIdx.x >> 5;
    const int mtile = blockIdx.x;                // 0..6249
    const int mrow  = mtile * 16 + (lane & 15);  // A row for this lane
    const int khalf = (lane >> 4) * 8;           // 0 or 8 (ISA 16-bit A layout)
    const int nt0   = wave * 2, nt1 = wave * 2 + 1;
    const int col0  = nt0 * 16 + (lane & 15);
    const int col1  = nt1 * 16 + (lane & 15);

    v8f acc0 = {}; v8f acc1 = {};

    #pragma unroll
    for (int ks = 0; ks < 4; ++ks) {             // K = 128 in steps of 32
        const int kbase = ks * 32;
        v16bf a, b0, b1;
        #pragma unroll
        for (int v = 0; v < 8; ++v) {
            // per-lane K index pattern (ISA 7.12.2, 16-bit A 16x32):
            // VGPR 0..3 -> K = khalf + 2v + p ; VGPR 4..7 -> K = 16 + khalf + 2(v-4) + p
            const int k0 = (v < 4) ? (khalf + v * 2) : (16 + khalf + (v - 4) * 2);
            #pragma unroll
            for (int p = 0; p < 2; ++p) {
                const int k = kbase + k0 + p;
                a [v * 2 + p] = (__bf16)x[(size_t)mrow * IN_DIM + k];
                b0[v * 2 + p] = (__bf16)W[(size_t)k * HC + col0];
                b1[v * 2 + p] = (__bf16)W[(size_t)k * HC + col1];
            }
        }
        acc0 = __builtin_amdgcn_wmma_f32_16x16x32_bf16(false, a, false, b0,
                                                       (short)0, acc0, false, false);
        acc1 = __builtin_amdgcn_wmma_f32_16x16x32_bf16(false, a, false, b1,
                                                       (short)0, acc1, false, false);
    }

    // C/D layout: VGPR r, lane L -> M = r + (L>>4)*8, Ncol = L&15
    #pragma unroll
    for (int r = 0; r < 8; ++r) {
        const int orow = mtile * 16 + r + ((lane >> 4) * 8);
        xl[(size_t)orow * HC + col0] = (__bf16)acc0[r];
        xl[(size_t)orow * HC + col1] = (__bf16)acc1[r];
    }
}

// ---------------------------------------------------------------- attention logits
__global__ void gat_logits_kernel(const __bf16* __restrict__ xl,
                                  const float* __restrict__ att_src,
                                  const float* __restrict__ att_dst,
                                  float* __restrict__ a_src,
                                  float* __restrict__ a_dst) {
    int t = blockIdx.x * blockDim.x + threadIdx.x;   // n*HEADS + h
    if (t >= NN * HEADS) return;
    const int n = t >> 2, h = t & 3;
    const __bf16* xr = xl + (size_t)n * HC + h * OUT_DIM;
    float s = 0.0f, d = 0.0f;
    #pragma unroll 8
    for (int c = 0; c < OUT_DIM; ++c) {
        const float v = (float)xr[c];
        s += v * att_src[h * OUT_DIM + c];
        d += v * att_dst[h * OUT_DIM + c];
    }
    a_src[t] = s;
    a_dst[t] = d;
}

// ---------------------------------------------------------------- segment max
__global__ void gat_edge_max_kernel(const long long* __restrict__ ei,
                                    const float* __restrict__ a_src,
                                    const float* __restrict__ a_dst,
                                    float* __restrict__ amax) {
    int e = blockIdx.x * blockDim.x + threadIdx.x;
    if (e >= TOT_E) return;
    int src, dst;
    if (e < NE) { src = (int)ei[e]; dst = (int)ei[NE + e]; }
    else        { src = dst = e - NE; }
    #pragma unroll
    for (int h = 0; h < HEADS; ++h) {
        const float a = lrelu(a_src[src * HEADS + h] + a_dst[dst * HEADS + h]);
        atomicMaxFloat(&amax[dst * HEADS + h], a);
    }
}

// ---------------------------------------------------------------- segment sum of exp
__global__ void gat_edge_sum_kernel(const long long* __restrict__ ei,
                                    const float* __restrict__ a_src,
                                    const float* __restrict__ a_dst,
                                    const float* __restrict__ amax,
                                    float* __restrict__ denom) {
    int e = blockIdx.x * blockDim.x + threadIdx.x;
    if (e >= TOT_E) return;
    int src, dst;
    if (e < NE) { src = (int)ei[e]; dst = (int)ei[NE + e]; }
    else        { src = dst = e - NE; }
    #pragma unroll
    for (int h = 0; h < HEADS; ++h) {
        const float a  = lrelu(a_src[src * HEADS + h] + a_dst[dst * HEADS + h]);
        const float ex = __expf(a - amax[dst * HEADS + h]);
        atomicAdd(&denom[dst * HEADS + h], ex);
    }
}

// ---------------------------------------------------------------- message passing
// 64 threads per edge (one per output channel); head-mean fused (x0.25 in LN).
__global__ void __launch_bounds__(256)
gat_edge_msg_kernel(const long long* __restrict__ ei,
                    const __bf16* __restrict__ xl,
                    const float* __restrict__ a_src,
                    const float* __restrict__ a_dst,
                    const float* __restrict__ amax,
                    const float* __restrict__ denom,
                    float* __restrict__ accum) {
    const long long t = (long long)blockIdx.x * blockDim.x + threadIdx.x;
    const int e = (int)(t >> 6);
    const int c = (int)(t & 63);
    if (e >= TOT_E) return;
    int src, dst;
    if (e < NE) { src = (int)ei[e]; dst = (int)ei[NE + e]; }
    else        { src = dst = e - NE; }

    float msum = 0.0f;
    #pragma unroll
    for (int h = 0; h < HEADS; ++h) {
        const float a  = lrelu(a_src[src * HEADS + h] + a_dst[dst * HEADS + h]);
        const float al = __expf(a - amax[dst * HEADS + h]) /
                         (denom[dst * HEADS + h] + 1e-16f);
        msum += al * (float)xl[(size_t)src * HC + h * OUT_DIM + c];
    }
    atomicAdd(&accum[(size_t)dst * OUT_DIM + c], msum);
}

// ---------------------------------------------------------------- head-mean + bias + LayerNorm
// one wave32 per node; lane handles channels {lane, lane+32}
__global__ void __launch_bounds__(256)
gat_ln_kernel(const float* __restrict__ accum,
              const float* __restrict__ bias,
              const float* __restrict__ gamma,
              const float* __restrict__ beta,
              float* __restrict__ out) {
    const int lane = threadIdx.x & 31;
    const int wave = threadIdx.x >> 5;
    const int node = blockIdx.x * 8 + wave;
    if (node >= NN) return;
    const float* row = accum + (size_t)node * OUT_DIM;
    float x0 = row[lane]      * 0.25f + bias[lane];
    float x1 = row[lane + 32] * 0.25f + bias[lane + 32];
    float s = x0 + x1;
    float q = x0 * x0 + x1 * x1;
    #pragma unroll
    for (int off = 16; off > 0; off >>= 1) {
        s += __shfl_xor(s, off, 32);
        q += __shfl_xor(q, off, 32);
    }
    const float mu  = s * (1.0f / 64.0f);
    const float var = q * (1.0f / 64.0f) - mu * mu;
    const float inv = rsqrtf(var + LN_EPS);
    float* orow = out + (size_t)node * OUT_DIM;
    orow[lane]      = (x0 - mu) * inv * gamma[lane]      + beta[lane];
    orow[lane + 32] = (x1 - mu) * inv * gamma[lane + 32] + beta[lane + 32];
}

// ---------------------------------------------------------------- launcher
extern "C" void kernel_launch(void* const* d_in, const int* in_sizes, int n_in,
                              void* d_out, int out_size, void* d_ws, size_t ws_size,
                              hipStream_t stream) {
    const float*     x       = (const float*)d_in[0];      // (NN,128)
    const long long* ei      = (const long long*)d_in[1];  // (2,NE) int64
    const float*     W       = (const float*)d_in[2];      // (128,256)
    const float*     att_src = (const float*)d_in[3];      // (4,64)
    const float*     att_dst = (const float*)d_in[4];      // (4,64)
    const float*     bias    = (const float*)d_in[5];      // (64,)
    const float*     gamma   = (const float*)d_in[6];      // (64,)
    const float*     beta    = (const float*)d_in[7];      // (64,)
    float*           out     = (float*)d_out;              // (NN,64)

    // workspace carve-out (~83 MB)
    char* p = (char*)d_ws;
    __bf16* xl    = (__bf16*)p; p += (size_t)NN * HC * sizeof(__bf16);   // 51.2 MB
    float*  a_src = (float*)p;  p += (size_t)NN * HEADS * sizeof(float); // 1.6 MB
    float*  a_dst = (float*)p;  p += (size_t)NN * HEADS * sizeof(float);
    float*  amax  = (float*)p;  p += (size_t)NN * HEADS * sizeof(float);
    float*  denom = (float*)p;  p += (size_t)NN * HEADS * sizeof(float);
    float*  accum = (float*)p;  p += (size_t)NN * OUT_DIM * sizeof(float); // 25.6 MB
    (void)ws_size; (void)in_sizes; (void)n_in; (void)out_size;

    // 1) init accumulators
    gat_init_kernel<<<(NN * OUT_DIM + 255) / 256, 256, 0, stream>>>(accum, amax, denom);

    // 2) WMMA GEMM: xl = bf16(x @ W)   (6250 M-tiles of 16 rows)
    gat_gemm_kernel<<<NN / 16, 256, 0, stream>>>(x, W, xl);

    // 3) attention logits
    gat_logits_kernel<<<(NN * HEADS + 255) / 256, 256, 0, stream>>>(
        xl, att_src, att_dst, a_src, a_dst);

    // 4) segment max
    gat_edge_max_kernel<<<(TOT_E + 255) / 256, 256, 0, stream>>>(ei, a_src, a_dst, amax);

    // 5) segment sum of exp
    gat_edge_sum_kernel<<<(TOT_E + 255) / 256, 256, 0, stream>>>(
        ei, a_src, a_dst, amax, denom);

    // 6) weighted scatter (64 threads per edge)
    {
        const long long tot = (long long)TOT_E * OUT_DIM;
        const int blocks = (int)((tot + 255) / 256);
        gat_edge_msg_kernel<<<blocks, 256, 0, stream>>>(
            ei, xl, a_src, a_dst, amax, denom, accum);
    }

    // 7) head-mean + bias + LayerNorm -> d_out
    gat_ln_kernel<<<(NN + 7) / 8, 256, 0, stream>>>(accum, bias, gamma, beta, out);
}
